// SwinTransformerTimm_73589969650208
// MI455X (gfx1250) — matrix operand order repaired and checked by reading, not compiled, not run on hardware
//
#include <hip/hip_runtime.h>

// ---------------------------------------------------------------------------
// Fused 14-head MLP: y_c = L4(relu(L3(relu(L2(relu(L1_c(x)))))))
// bf16 V_WMMA_F32_16X16X32_BF16 for layers 1-3 (f32 accumulate), fully fused
// so h1 [14,16384,384] never touches HBM. The 128x768 bf16 x-tile is staged
// into LDS once per block with the Tensor Data Mover (tensor_load_to_lds),
// so A-fragments come from LDS and x leaves L2 exactly once per block.
// ---------------------------------------------------------------------------

typedef __attribute__((ext_vector_type(16))) __bf16        bf16x16;
typedef __attribute__((ext_vector_type(8)))  float         f32x8;
typedef __attribute__((ext_vector_type(4)))  unsigned int  uint32x4;
typedef __attribute__((ext_vector_type(8)))  unsigned int  uint32x8;

union Frag32B {                 // one WMMA 16-bit A/B operand: 32 bytes/lane
  bf16x16 v;
  uint4   q[2];
};

__device__ __forceinline__ unsigned short f2bf(float f) {  // RNE f32->bf16
  unsigned u = __builtin_bit_cast(unsigned, f);
  u += 0x7FFFu + ((u >> 16) & 1u);
  return (unsigned short)(u >> 16);
}

__device__ __forceinline__ f32x8 wmma_bf16(bf16x16 a, bf16x16 b, f32x8 c) {
  return __builtin_amdgcn_wmma_f32_16x16x32_bf16(false, a, false, b,
                                                 (short)0, c, false, false);
}

__device__ __forceinline__ Frag32B load_frag(const unsigned short* p0,
                                             const unsigned short* p1) {
  Frag32B f;
  f.q[0] = *reinterpret_cast<const uint4*>(p0);
  f.q[1] = *reinterpret_cast<const uint4*>(p1);
  return f;
}

// ------------------------------ constants ----------------------------------
#define NB    16384
#define NF    768
#define NC    14
#define NH1   384
#define NH2   48
#define NOUT  2
#define BM    128        // rows per block
#define BLOCK 256        // 8 waves (wave32)

// ------------------- fp32 -> bf16 pre-conversion kernel --------------------
__global__ __launch_bounds__(256)
void cvt_f32_to_bf16(const float* __restrict__ in,
                     unsigned short* __restrict__ out, int n) {
  int i = (blockIdx.x * 256 + threadIdx.x) * 4;   // n is a multiple of 4
  if (i >= n) return;
  float4 f = *reinterpret_cast<const float4*>(in + i);
  uint2 pk;
  pk.x = (unsigned)f2bf(f.x) | ((unsigned)f2bf(f.y) << 16);
  pk.y = (unsigned)f2bf(f.z) | ((unsigned)f2bf(f.w) << 16);
  *reinterpret_cast<uint2*>(out + i) = pk;
}

// ------------------------------ fused kernel -------------------------------
__global__ __launch_bounds__(BLOCK)
void mlp_heads_fused(const unsigned short* __restrict__ xb,   // [B][768] bf16
                     const unsigned short* __restrict__ w1b,  // [C][384][768] bf16
                     const float* __restrict__ b1,            // [C][384]
                     const unsigned short* __restrict__ w2b,  // [C][48][384] bf16
                     const float* __restrict__ b2,            // [C][48]
                     const float* __restrict__ W3,            // [C][48][48] f32
                     const float* __restrict__ b3,            // [C][48]
                     const float* __restrict__ W4,            // [C][2][48] f32
                     const float* __restrict__ b4,            // [C][2]
                     float* __restrict__ out)                 // [C][B][2]
{
  const int c    = blockIdx.y;
  const int b0   = blockIdx.x * BM;
  const int tid  = threadIdx.x;
  const int wave = tid >> 5;
  const int lane = tid & 31;
  const int lh   = lane & 15;          // N (or M) index within fragment
  const int hi   = (lane >> 4) & 1;    // lane-half selects K sub-chunk

  // Phase 1 (x-tile) and phase 2 (h2/h3) have disjoint lifetimes -> overlap.
  __shared__ __align__(16) union {
    unsigned short sX[BM][NF];                   // 192 KB: bf16 x-tile
    struct {
      unsigned short h2[BM][64];                 // 16 KB, K padded to 64
      float          h3[BM][48];                 // 24 KB
    } p2;
  } U;
  __shared__ __align__(16) unsigned short sStage[8][16][32]; // L1->L2 bounce
  __shared__ __align__(16) unsigned short sW3[48][64];       // K padded to 64

  // ---- TDM: DMA the 128x768 bf16 x-tile into LDS (one descriptor) --------
  if (wave == 0) {
    unsigned           ldsa = (unsigned)(size_t)(void*)U.sX;
    unsigned long long ga   = (unsigned long long)(size_t)(xb + (size_t)b0 * NF);
    uint32x4 g0;
    g0[0] = 1u;                                        // count=1 (valid D#)
    g0[1] = ldsa;                                      // lds_addr
    g0[2] = (unsigned)(ga & 0xFFFFFFFFu);              // global_addr[31:0]
    g0[3] = (unsigned)((ga >> 32) & 0x1FFFFFFu)        // global_addr[56:32]
            | (2u << 30);                              // type=2 (image)
    uint32x8 g1;
    g1[0] = 1u << 16;                                  // data_size=1 -> 2 bytes
    g1[1] = ((unsigned)NF & 0xFFFFu) << 16;            // tensor_dim0[15:0]=768
    g1[2] = ((unsigned)NB & 0xFFFFu) << 16;            // tensor_dim1[15:0]=16384
    g1[3] = ((unsigned)NF & 0xFFFFu) << 16;            // tile_dim0=768
    g1[4] = (unsigned)BM;                              // tile_dim1=128, tile_dim2=0
    g1[5] = (unsigned)NF;                              // tensor_dim0_stride=768
    g1[6] = 0u;
    g1[7] = 0u;
    asm volatile("tensor_load_to_lds %0, %1" :: "s"(g0), "s"(g1) : "memory");
  }

  // Stage W3 (bf16, zero-padded K 48->64) while the TDM streams x.
  for (int i = tid; i < 48 * 64; i += BLOCK) {
    int n = i >> 6, k = i & 63;
    sW3[n][k] = (k < 48) ? f2bf(W3[((size_t)c * 48 + n) * 48 + k]) : (unsigned short)0;
  }
  if (wave == 0) __builtin_amdgcn_s_wait_tensorcnt(0);
  __syncthreads();                                     // x-tile + W3 visible

  const int m0 = wave * 16;            // this wave's M-tile inside the block
  const f32x8 vzero = {0.f, 0.f, 0.f, 0.f, 0.f, 0.f, 0.f, 0.f};
  const unsigned short* xrow = &U.sX[m0 + lh][0];                 // A row = M
  const unsigned short* w1c  = w1b + (size_t)c * NH1 * NF;
  const unsigned short* w2c  = w2b + (size_t)c * NH2 * NH1;

  // ---------------- Layer 1 fused into Layer 2 -----------------------------
  // Layer-2 K dim == Layer-1 N dim: produce h1 32 columns at a time and
  // immediately fold them into the persistent layer-2 accumulators.
  f32x8 acc2[3];                               // 16 x 48 output of layer 2
  #pragma unroll
  for (int t = 0; t < 3; ++t) acc2[t] = vzero;

  for (int g = 0; g < NH1 / 32; ++g) {         // 12 groups of 32 h1 columns
    f32x8 acc1[2] = {vzero, vzero};
    for (int k0 = 0; k0 < NF; k0 += 32) {      // K = 768, A from LDS
      Frag32B A = load_frag(xrow + k0 + hi * 8, xrow + k0 + hi * 8 + 16);
      #pragma unroll
      for (int t = 0; t < 2; ++t) {
        const unsigned short* bp =
            w1c + (size_t)(g * 32 + t * 16 + lh) * NF + k0 + hi * 8;
        Frag32B Bf = load_frag(bp, bp + 16);
        acc1[t] = wmma_bf16(A.v, Bf.v, acc1[t]);
      }
    }
    // bias + relu + bf16, bounce through wave-private LDS (D->A transpose)
    #pragma unroll
    for (int t = 0; t < 2; ++t) {
      float bias = b1[(size_t)c * NH1 + g * 32 + t * 16 + lh];
      #pragma unroll
      for (int r = 0; r < 8; ++r) {
        float v = acc1[t][r] + bias;
        sStage[wave][r + hi * 8][t * 16 + lh] = f2bf(v > 0.f ? v : 0.f);
      }
    }
    const unsigned short* a2p = &sStage[wave][lh][hi * 8];
    Frag32B A2 = load_frag(a2p, a2p + 16);
    #pragma unroll
    for (int t = 0; t < 3; ++t) {
      const unsigned short* bp =
          w2c + (size_t)(t * 16 + lh) * NH1 + g * 32 + hi * 8;
      Frag32B B2 = load_frag(bp, bp + 16);
      acc2[t] = wmma_bf16(A2.v, B2.v, acc2[t]);
    }
  }

  __syncthreads();   // all sX reads done; safe to reuse that LDS for h2/h3

  // Zero-pad this wave's h2 columns 48..63 (wave-private rows).
  for (int i = lane; i < 16 * 16; i += 32)
    U.p2.h2[m0 + (i >> 4)][48 + (i & 15)] = 0;

  // Layer-2 epilogue -> h2 (bf16, wave-private rows)
  #pragma unroll
  for (int t = 0; t < 3; ++t) {
    float bias = b2[(size_t)c * NH2 + t * 16 + lh];
    #pragma unroll
    for (int r = 0; r < 8; ++r) {
      float v = acc2[t][r] + bias;
      U.p2.h2[m0 + r + hi * 8][t * 16 + lh] = f2bf(v > 0.f ? v : 0.f);
    }
  }

  // ---------------- Layer 3: [16x48] = [16x64] x [64x48] -------------------
  f32x8 acc3[3];
  #pragma unroll
  for (int t = 0; t < 3; ++t) acc3[t] = vzero;
  #pragma unroll
  for (int k0 = 0; k0 < 64; k0 += 32) {
    const unsigned short* a3p = &U.p2.h2[m0 + lh][k0 + hi * 8];
    Frag32B A3 = load_frag(a3p, a3p + 16);
    #pragma unroll
    for (int t = 0; t < 3; ++t) {
      const unsigned short* bp = &sW3[t * 16 + lh][k0 + hi * 8];
      Frag32B B3 = load_frag(bp, bp + 16);
      acc3[t] = wmma_bf16(A3.v, B3.v, acc3[t]);
    }
  }
  #pragma unroll
  for (int t = 0; t < 3; ++t) {
    float bias = b3[(size_t)c * NH2 + t * 16 + lh];
    #pragma unroll
    for (int r = 0; r < 8; ++r) {
      float v = acc3[t][r] + bias;
      U.p2.h3[m0 + r + hi * 8][t * 16 + lh] = v > 0.f ? v : 0.f;
    }
  }
  __syncthreads();

  // ---------------- Layer 4: tiny (OUT=2, K=48), plain VALU ----------------
  {
    int r = tid >> 1, o = tid & 1;               // 128 rows x 2 outputs
    const float* w4 = W4 + ((size_t)c * NOUT + o) * NH2;
    float a = b4[(size_t)c * NOUT + o];
    #pragma unroll 4
    for (int k = 0; k < NH2; ++k) a = fmaf(U.p2.h3[r][k], w4[k], a);
    out[((size_t)c * NB + b0 + r) * NOUT + o] = a;
  }
}

// ------------------------------- launcher ----------------------------------
extern "C" void kernel_launch(void* const* d_in, const int* in_sizes, int n_in,
                              void* d_out, int out_size, void* d_ws, size_t ws_size,
                              hipStream_t stream) {
  const float* x  = (const float*)d_in[0];
  const float* W1 = (const float*)d_in[1];
  const float* b1 = (const float*)d_in[2];
  const float* W2 = (const float*)d_in[3];
  const float* b2 = (const float*)d_in[4];
  const float* W3 = (const float*)d_in[5];
  const float* b3 = (const float*)d_in[6];
  const float* W4 = (const float*)d_in[7];
  const float* b4 = (const float*)d_in[8];
  float* out = (float*)d_out;

  const size_t nX  = (size_t)NB * NF;          // 12,582,912
  const size_t nW1 = (size_t)NC * NH1 * NF;    //  4,128,768
  const size_t nW2 = (size_t)NC * NH2 * NH1;   //    258,048

  unsigned short* xb  = (unsigned short*)d_ws;
  unsigned short* w1b = xb  + nX;
  unsigned short* w2b = w1b + nW1;             // ~34 MB of workspace total

  auto blocks4 = [](size_t n) { return (unsigned)((n / 4 + 255) / 256); };
  cvt_f32_to_bf16<<<blocks4(nX),  256, 0, stream>>>(x,  xb,  (int)nX);
  cvt_f32_to_bf16<<<blocks4(nW1), 256, 0, stream>>>(W1, w1b, (int)nW1);
  cvt_f32_to_bf16<<<blocks4(nW2), 256, 0, stream>>>(W2, w2b, (int)nW2);

  dim3 grid(NB / BM, NC);                      // 128 x 14 blocks
  mlp_heads_fused<<<grid, BLOCK, 0, stream>>>(xb, w1b, b1, w2b, b2,
                                              W3, b3, W4, b4, out);
}
